// RNN_48610439856342
// MI455X (gfx1250) — matrix operand order repaired and checked by reading, compiled, not run on hardware
//
#include <hip/hip_runtime.h>
#include <hip/hip_bf16.h>
#include <stdint.h>

// Problem dims (fixed by the reference)
#define L_SEQ 256
#define BATCH 128
#define DIM_D 2048
#define DIM_H 2048

typedef __attribute__((ext_vector_type(16))) __bf16 bf16x16;
typedef __attribute__((ext_vector_type(8)))  float  f32x8;

union FragBF {
    uint4   q[2];   // 32 bytes = 16 bf16
    bf16x16 v;
};

__device__ __forceinline__ uint16_t f2bf(float f) {
    union { float f; uint32_t u; } x; x.f = f;
    uint32_t u = x.u;
    u += 0x7fffu + ((u >> 16) & 1u);   // round-to-nearest-even
    return (uint16_t)(u >> 16);
}

__device__ __forceinline__ uint32_t pk2(float lo, float hi) {
    return (uint32_t)f2bf(lo) | ((uint32_t)f2bf(hi) << 16);
}

// ---------------------------------------------------------------------------
// f32 [K x N] -> bf16 transposed [N x K]   (weights, one-time per call)
// ---------------------------------------------------------------------------
__global__ __launch_bounds__(256)
void cvt_transpose_kernel(const float* __restrict__ W, uint16_t* __restrict__ WT,
                          int K, int N) {
    int idx = blockIdx.x * 256 + threadIdx.x;
    int k = idx / N;
    int n = idx - k * N;
    WT[(size_t)n * K + k] = f2bf(W[idx]);
}

// f32 -> bf16, straight copy, 8 elements/thread (for the big input tensor)
__global__ __launch_bounds__(256)
void cvt_bf16_kernel(const float* __restrict__ X, uint16_t* __restrict__ Y) {
    size_t i = ((size_t)blockIdx.x * 256 + threadIdx.x) * 8;
    float4 x0 = *(const float4*)(X + i);
    float4 x1 = *(const float4*)(X + i + 4);
    *(uint4*)(Y + i) = make_uint4(pk2(x0.x, x0.y), pk2(x0.z, x0.w),
                                  pk2(x1.x, x1.y), pk2(x1.z, x1.w));
}

__global__ __launch_bounds__(256)
void zero_bf16_kernel(uint16_t* __restrict__ p) {
    p[blockIdx.x * 256 + threadIdx.x] = 0;
}

// ---------------------------------------------------------------------------
// Phase 1: xp = inputs @ W_ih + b
// A: [M x K] (f32 or pre-converted bf16), BT: [N x K] bf16, C: [M x N] f32
// 256 threads = 8 waves; WG tile 64(M) x 128(N); wave tile 16 x 64.
// K = DIM_D, N = DIM_H are compile-time so all strides fold into immediates.
// ---------------------------------------------------------------------------
template<typename AT>
__global__ __launch_bounds__(256)
void rnn_xproj_kernel(const AT* __restrict__ A,
                      const uint16_t* __restrict__ BT,
                      const float* __restrict__ bias,
                      float* __restrict__ C) {
    constexpr int K = DIM_D;
    constexpr int N = DIM_H;
    const int lane  = threadIdx.x & 31;
    const int wave  = threadIdx.x >> 5;
    const int waveM = wave >> 1;          // 0..3
    const int waveN = wave & 1;           // 0..1
    const int half  = lane >> 4;          // 0..1
    const int ln    = lane & 15;          // 0..15

    const int rowb = blockIdx.y * 64 + waveM * 16;
    const int colb = blockIdx.x * 128 + waveN * 64;

    const AT* arow = A + (size_t)(rowb + ln) * K;

    const uint16_t* brow0 = BT + (size_t)(colb +  0 + ln) * K + half * 16;
    const uint16_t* brow1 = BT + (size_t)(colb + 16 + ln) * K + half * 16;
    const uint16_t* brow2 = BT + (size_t)(colb + 32 + ln) * K + half * 16;
    const uint16_t* brow3 = BT + (size_t)(colb + 48 + ln) * K + half * 16;

    f32x8 acc0 = {}, acc1 = {}, acc2 = {}, acc3 = {};

    for (int kk = 0; kk < K; kk += 32) {
        // A fragment (bf16 16x32 layout): per lane k = kk+half*16..+15,
        // contiguous (matches B).
        FragBF aF;
        if constexpr (sizeof(AT) == 4) {           // f32 source, convert in-flight
            const float* ap = (const float*)arow + kk + half * 16;
            __builtin_prefetch(ap + 64, 0, 3);
            float4 a0 = *(const float4*)(ap);
            float4 a1 = *(const float4*)(ap + 4);
            float4 a2 = *(const float4*)(ap + 8);
            float4 a3 = *(const float4*)(ap + 12);
            aF.q[0] = make_uint4(pk2(a0.x, a0.y), pk2(a0.z, a0.w),
                                 pk2(a1.x, a1.y), pk2(a1.z, a1.w));
            aF.q[1] = make_uint4(pk2(a2.x, a2.y), pk2(a2.z, a2.w),
                                 pk2(a3.x, a3.y), pk2(a3.z, a3.w));
        } else {                                    // pre-converted bf16 source
            const uint16_t* ap = (const uint16_t*)arow + kk + half * 16;
            __builtin_prefetch(ap + 128, 0, 3);
            aF.q[0] = *(const uint4*)(ap);
            aF.q[1] = *(const uint4*)(ap + 8);
        }

        // B fragments (bf16 32x16 layout): per lane k = kk+half*16..+15,
        // contiguous along K in the transposed weights.
        FragBF b0, b1, b2, b3;
        b0.q[0] = *(const uint4*)(brow0 + kk);  b0.q[1] = *(const uint4*)(brow0 + kk + 8);
        b1.q[0] = *(const uint4*)(brow1 + kk);  b1.q[1] = *(const uint4*)(brow1 + kk + 8);
        b2.q[0] = *(const uint4*)(brow2 + kk);  b2.q[1] = *(const uint4*)(brow2 + kk + 8);
        b3.q[0] = *(const uint4*)(brow3 + kk);  b3.q[1] = *(const uint4*)(brow3 + kk + 8);
        __builtin_prefetch(brow0 + kk + 32, 0, 3);

        acc0 = __builtin_amdgcn_wmma_f32_16x16x32_bf16(false, aF.v, false, b0.v, (short)0, acc0, false, false);
        acc1 = __builtin_amdgcn_wmma_f32_16x16x32_bf16(false, aF.v, false, b1.v, (short)0, acc1, false, false);
        acc2 = __builtin_amdgcn_wmma_f32_16x16x32_bf16(false, aF.v, false, b2.v, (short)0, acc2, false, false);
        acc3 = __builtin_amdgcn_wmma_f32_16x16x32_bf16(false, aF.v, false, b3.v, (short)0, acc3, false, false);
    }

    // Epilogue: C layout VGPR v -> row (v + 8*half), col ln. One 64-bit base,
    // all remaining offsets are compile-time immediates.
    f32x8 accs[4] = { acc0, acc1, acc2, acc3 };
    float* cp = C + (size_t)(rowb + 8 * half) * N + colb + ln;
#pragma unroll
    for (int nt = 0; nt < 4; ++nt) {
        const float bv = bias[colb + nt * 16 + ln];
#pragma unroll
        for (int v = 0; v < 8; ++v)
            cp[(size_t)v * N + nt * 16] = accs[nt][v] + bv;
    }
}

// ---------------------------------------------------------------------------
// Phase 2 step: h_new = tanh(xp_t + h_prev @ W_hh)
// xio: [B x H] f32 holds xp_t, overwritten in place with h_t (same thread
// reads then writes each element). hprev/hnext: bf16 [B x H].
// WT: [H x H] bf16 transposed. Grid: (H/128, B/64) = (16, 2).
// ---------------------------------------------------------------------------
__global__ __launch_bounds__(256)
void rnn_step_kernel(float* xio,
                     const uint16_t* __restrict__ hprev,
                     const uint16_t* __restrict__ WT,
                     uint16_t* __restrict__ hnext) {
    constexpr int K = DIM_H;
    constexpr int N = DIM_H;
    const int lane  = threadIdx.x & 31;
    const int wave  = threadIdx.x >> 5;
    const int waveM = wave >> 1;
    const int waveN = wave & 1;
    const int half  = lane >> 4;
    const int ln    = lane & 15;

    const int rowb = blockIdx.y * 64 + waveM * 16;
    const int colb = blockIdx.x * 128 + waveN * 64;

    const uint16_t* arow = hprev + (size_t)(rowb + ln) * K;

    const uint16_t* brow0 = WT + (size_t)(colb +  0 + ln) * K + half * 16;
    const uint16_t* brow1 = WT + (size_t)(colb + 16 + ln) * K + half * 16;
    const uint16_t* brow2 = WT + (size_t)(colb + 32 + ln) * K + half * 16;
    const uint16_t* brow3 = WT + (size_t)(colb + 48 + ln) * K + half * 16;

    f32x8 acc0 = {}, acc1 = {}, acc2 = {}, acc3 = {};

    for (int kk = 0; kk < K; kk += 32) {
        FragBF aF;
        const uint16_t* ap = arow + kk + half * 16;
        aF.q[0] = *(const uint4*)(ap);        // k = kk+half*16   .. +7
        aF.q[1] = *(const uint4*)(ap + 8);    // k = kk+half*16+8 .. +15

        FragBF b0, b1, b2, b3;
        b0.q[0] = *(const uint4*)(brow0 + kk);  b0.q[1] = *(const uint4*)(brow0 + kk + 8);
        b1.q[0] = *(const uint4*)(brow1 + kk);  b1.q[1] = *(const uint4*)(brow1 + kk + 8);
        b2.q[0] = *(const uint4*)(brow2 + kk);  b2.q[1] = *(const uint4*)(brow2 + kk + 8);
        b3.q[0] = *(const uint4*)(brow3 + kk);  b3.q[1] = *(const uint4*)(brow3 + kk + 8);

        acc0 = __builtin_amdgcn_wmma_f32_16x16x32_bf16(false, aF.v, false, b0.v, (short)0, acc0, false, false);
        acc1 = __builtin_amdgcn_wmma_f32_16x16x32_bf16(false, aF.v, false, b1.v, (short)0, acc1, false, false);
        acc2 = __builtin_amdgcn_wmma_f32_16x16x32_bf16(false, aF.v, false, b2.v, (short)0, acc2, false, false);
        acc3 = __builtin_amdgcn_wmma_f32_16x16x32_bf16(false, aF.v, false, b3.v, (short)0, acc3, false, false);
    }

    f32x8 accs[4] = { acc0, acc1, acc2, acc3 };
    const size_t base = (size_t)(rowb + 8 * half) * N + colb + ln;
    float*    xp = xio + base;
    uint16_t* hp = hnext + base;
#pragma unroll
    for (int nt = 0; nt < 4; ++nt) {
#pragma unroll
        for (int v = 0; v < 8; ++v) {
            const size_t off = (size_t)v * N + nt * 16;
            const float hv = tanhf(accs[nt][v] + xp[off]);
            xp[off] = hv;              // outputs[t] = h_t (in place over xp_t)
            hp[off] = f2bf(hv);        // bf16 copy for next step's A operand
        }
    }
}

// ---------------------------------------------------------------------------
extern "C" void kernel_launch(void* const* d_in, const int* in_sizes, int n_in,
                              void* d_out, int out_size, void* d_ws, size_t ws_size,
                              hipStream_t stream) {
    (void)in_sizes; (void)n_in; (void)out_size;

    const float* inputs = (const float*)d_in[0];   // [L, B, D]
    const float* W_ih   = (const float*)d_in[1];   // [D, H]
    const float* W_hh   = (const float*)d_in[2];   // [H, H]
    const float* bias   = (const float*)d_in[3];   // [H]
    float* out = (float*)d_out;                    // [L*B*H] outputs ++ [B*H] final

    const int M1 = L_SEQ * BATCH;                  // 32768

    char* ws = (char*)d_ws;
    const size_t wih_bytes = (size_t)DIM_D * DIM_H * 2;     // 8 MB
    const size_t whh_bytes = (size_t)DIM_H * DIM_H * 2;     // 8 MB
    const size_t h_bytes   = (size_t)BATCH * DIM_H * 2;     // 512 KB
    uint16_t* WT_ih = (uint16_t*)ws;
    uint16_t* WT_hh = (uint16_t*)(ws + wih_bytes);
    uint16_t* hbuf0 = (uint16_t*)(ws + wih_bytes + whh_bytes);
    uint16_t* hbuf1 = hbuf0 + (size_t)BATCH * DIM_H;
    const size_t base_bytes = wih_bytes + whh_bytes + 2 * h_bytes;   // ~17 MB
    uint16_t* A16 = (uint16_t*)(ws + base_bytes);
    const size_t a16_bytes = (size_t)M1 * DIM_D * 2;                  // 128 MB

    // ws_size is constant per run -> this branch is deterministic.
    const bool preconvert = (ws_size >= base_bytes + a16_bytes);

    // Weight prep (runs every call; ws contents are never assumed preserved).
    cvt_transpose_kernel<<<(DIM_D * DIM_H) / 256, 256, 0, stream>>>(W_ih, WT_ih, DIM_D, DIM_H);
    cvt_transpose_kernel<<<(DIM_H * DIM_H) / 256, 256, 0, stream>>>(W_hh, WT_hh, DIM_H, DIM_H);
    zero_bf16_kernel<<<(BATCH * DIM_H) / 256, 256, 0, stream>>>(hbuf0);

    // Phase 1: xp -> outputs region of d_out.
    dim3 g1(DIM_H / 128, M1 / 64);                    // (16, 512)
    if (preconvert) {
        cvt_bf16_kernel<<<((size_t)M1 * DIM_D) / (256 * 8), 256, 0, stream>>>(inputs, A16);
        rnn_xproj_kernel<uint16_t><<<g1, 256, 0, stream>>>(A16, WT_ih, bias, out);
    } else {
        rnn_xproj_kernel<float><<<g1, 256, 0, stream>>>(inputs, WT_ih, bias, out);
    }

    // Phase 2: 256 dependent steps, in place over xp slices.
    dim3 g2(DIM_H / 128, BATCH / 64);                 // (16, 2)
    for (int t = 0; t < L_SEQ; ++t) {
        const uint16_t* hp = (t & 1) ? hbuf1 : hbuf0;
        uint16_t*       hn = (t & 1) ? hbuf0 : hbuf1;
        float* xo = out + (size_t)t * BATCH * DIM_H;
        rnn_step_kernel<<<g2, 256, 0, stream>>>(xo, hp, WT_hh, hn);
    }

    // final_state = h_{L-1} = outputs[L-1]
    hipMemcpyAsync(out + (size_t)L_SEQ * BATCH * DIM_H,
                   out + (size_t)(L_SEQ - 1) * BATCH * DIM_H,
                   (size_t)BATCH * DIM_H * sizeof(float),
                   hipMemcpyDeviceToDevice, stream);
}
